// Model_30253749633669
// MI455X (gfx1250) — compile-verified, matrix operand order checked by reading, not run on hardware
//
#include <hip/hip_runtime.h>
#include <hip/hip_bf16.h>

// ---------------------------------------------------------------------------
// Problem constants (match reference)
// ---------------------------------------------------------------------------
constexpr int BB   = 8;      // clouds
constexpr int NN   = 4096;   // points / cloud
constexpr int SS   = 256;    // FPS samples
constexpr int KNB  = 128;    // neighbor cap
constexpr float R2 = 0.2f * 0.2f;
constexpr int MG   = 45;
constexpr int GG   = MG * MG;      // 2025 grid points
constexpr int BOT  = 1024;         // bottleneck
constexpr int ROWS_DEC = 16208;    // B*G = 16200 padded to multiple of 16

typedef __attribute__((ext_vector_type(16))) _Float16 v16h;
typedef __attribute__((ext_vector_type(8)))  _Float16 v8h;
typedef __attribute__((ext_vector_type(8)))  float    v8f;

// ---------------------------------------------------------------------------
// WMMA fragment helpers (CDNA5 16x16x32 f16, wave32 layouts per ISA 7.12.2)
//   A (16x32 f16): lane L -> row = L%16; halves j: k = 8*(L/16) + j + (j>=8?8:0)
//   So per-lane: 8 halves at k-base, 8 halves at k-base+16 (two b128 loads).
//   B packed layout (prepacked): per (ktile,ntile): [lane][j] contiguous.
//   D (16x16 f32): elem r -> row = r + 8*(L/16), col = L%16.
// ---------------------------------------------------------------------------
__device__ __forceinline__ v16h load_afrag(const _Float16* p /* row base + kchunk */) {
    v8h lo = *(const v8h*)p;
    v8h hi = *(const v8h*)(p + 16);
    v16h a;
#pragma unroll
    for (int t = 0; t < 8; ++t) { a[t] = lo[t]; a[t + 8] = hi[t]; }
    return a;
}

__device__ __forceinline__ v8f zero_v8f() {
    v8f z;
#pragma unroll
    for (int t = 0; t < 8; ++t) z[t] = 0.0f;
    return z;
}

__device__ __forceinline__ float act_apply(float v, int act) {
    if (act == 1) return v > 0.0f ? v : 0.01f * v;   // leaky relu (slope 0.01)
    if (act == 2) return v > 0.0f ? v : 0.0f;        // relu
    return v;
}

// ---------------------------------------------------------------------------
// Weight packing: f32 row-major [K,N] -> f16 WMMA B-fragment layout, padded.
// dst element i: j = i&15, lane = (i>>4)&31, tile = i>>9 (= kt*nTiles+nt)
// ---------------------------------------------------------------------------
__global__ void packw_kernel(const float* __restrict__ W, int K, int Nn,
                             int Kp, int Np, _Float16* __restrict__ dst) {
    int i = blockIdx.x * blockDim.x + threadIdx.x;
    int total = Kp * Np;
    if (i >= total) return;
    int j    = i & 15;
    int lane = (i >> 4) & 31;
    int tile = i >> 9;
    int nTiles = Np >> 4;
    int kt = tile / nTiles, nt = tile % nTiles;
    int base = (lane < 16) ? 0 : 8;
    int k = kt * 32 + base + j + (j >= 8 ? 8 : 0);
    int n = nt * 16 + (lane & 15);
    float v = (k < K && n < Nn) ? W[(size_t)k * Nn + n] : 0.0f;
    dst[i] = (_Float16)v;
}

// ---------------------------------------------------------------------------
// Transformer: x = lrelu(pos @ Wt + bt)  [B*N,3]x[3,64] -> f16
// ---------------------------------------------------------------------------
__global__ void transformer_kernel(const float* __restrict__ pos,
                                   const float* __restrict__ Wt,
                                   const float* __restrict__ bt,
                                   _Float16* __restrict__ xh) {
    int i = blockIdx.x * blockDim.x + threadIdx.x;
    if (i >= BB * NN * 64) return;
    int c = i & 63;
    int p = i >> 6;
    const float* pp = pos + (size_t)p * 3;
    float v = pp[0] * Wt[c] + pp[1] * Wt[64 + c] + pp[2] * Wt[128 + c] + bt[c];
    v = v > 0.0f ? v : 0.01f * v;
    xh[i] = (_Float16)v;
}

// ---------------------------------------------------------------------------
// Farthest point sampling: one block per cloud, 1024 threads, 4 pts/thread.
// Argmax with first-index tie-break via packed (dist_bits<<32)|(N-1-idx).
// ---------------------------------------------------------------------------
__global__ __launch_bounds__(1024)
void fps_kernel(const float* __restrict__ pos, int* __restrict__ idx,
                float* __restrict__ new_pos) {
    int b = blockIdx.x;
    int tid = threadIdx.x;
    const float* P = pos + (size_t)b * NN * 3;
    float px[4], py[4], pz[4], mind[4];
#pragma unroll
    for (int i = 0; i < 4; ++i) {
        int p = tid * 4 + i;
        px[i] = P[p * 3 + 0];
        py[i] = P[p * 3 + 1];
        pz[i] = P[p * 3 + 2];
        mind[i] = 3.402823466e38f;
    }
    __shared__ float cur[3];
    __shared__ unsigned long long best;
    if (tid == 0) {
        idx[b * SS] = 0;
        new_pos[(size_t)(b * SS) * 3 + 0] = P[0];
        new_pos[(size_t)(b * SS) * 3 + 1] = P[1];
        new_pos[(size_t)(b * SS) * 3 + 2] = P[2];
        cur[0] = P[0]; cur[1] = P[1]; cur[2] = P[2];
        best = 0ull;
    }
    __syncthreads();
    for (int it = 1; it < SS; ++it) {
        float cx = cur[0], cy = cur[1], cz = cur[2];
        unsigned long long lkey = 0ull;
#pragma unroll
        for (int i = 0; i < 4; ++i) {
            float dx = px[i] - cx, dy = py[i] - cy, dz = pz[i] - cz;
            float d = dx * dx + dy * dy + dz * dz;
            mind[i] = fminf(mind[i], d);
            unsigned kb = __float_as_uint(mind[i]);
            unsigned long long kk =
                ((unsigned long long)kb << 32) | (unsigned)(NN - 1 - (tid * 4 + i));
            lkey = kk > lkey ? kk : lkey;
        }
#pragma unroll
        for (int off = 16; off > 0; off >>= 1) {
            unsigned lo = (unsigned)lkey, hi = (unsigned)(lkey >> 32);
            unsigned olo = __shfl_xor(lo, off);
            unsigned ohi = __shfl_xor(hi, off);
            unsigned long long o = ((unsigned long long)ohi << 32) | olo;
            lkey = o > lkey ? o : lkey;
        }
        if ((tid & 31) == 0) atomicMax(&best, lkey);
        __syncthreads();
        if (tid == 0) {
            unsigned long long k = best;
            int sel = NN - 1 - (int)(k & 0xFFFFFFFFull);
            idx[b * SS + it] = sel;
            float sx = P[sel * 3 + 0], sy = P[sel * 3 + 1], sz = P[sel * 3 + 2];
            size_t o = (size_t)(b * SS + it) * 3;
            new_pos[o + 0] = sx; new_pos[o + 1] = sy; new_pos[o + 2] = sz;
            cur[0] = sx; cur[1] = sy; cur[2] = sz;
            best = 0ull;
        }
        __syncthreads();
    }
}

// ---------------------------------------------------------------------------
// Ball query: one wave32 per (b,s); first <=128 in-ball indices in order.
// ---------------------------------------------------------------------------
__global__ void ballq_kernel(const float* __restrict__ pos,
                             const float* __restrict__ new_pos,
                             int* __restrict__ nidx, int* __restrict__ nvalid) {
    int gw = (blockIdx.x * blockDim.x + threadIdx.x) >> 5;
    if (gw >= BB * SS) return;
    int lane = threadIdx.x & 31;
    int b = gw >> 8;
    const float* P = pos + (size_t)b * NN * 3;
    float cx = new_pos[(size_t)gw * 3 + 0];
    float cy = new_pos[(size_t)gw * 3 + 1];
    float cz = new_pos[(size_t)gw * 3 + 2];
    int count = 0;
    int* out = nidx + (size_t)gw * KNB;
    for (int base = 0; base < NN && count < KNB; base += 32) {
        int j = base + lane;
        float dx = P[j * 3 + 0] - cx;
        float dy = P[j * 3 + 1] - cy;
        float dz = P[j * 3 + 2] - cz;
        float d = dx * dx + dy * dy + dz * dz;
        bool in = d <= R2;
        unsigned mask = __builtin_amdgcn_ballot_w32(in);
        int pre = __popc(mask & ((1u << lane) - 1u));
        if (in && count + pre < KNB) out[count + pre] = j;
        count = min(count + __popc(mask), KNB);
    }
    for (int k = count + lane; k < KNB; k += 32) out[k] = 0;
    if (lane == 0) nvalid[gw] = count;
}

// ---------------------------------------------------------------------------
// Fused PointConv: per (b,s) workgroup of 8 waves.
//   feat[128x96] -> h1[128x64] -> h2[128x128] -> (512 cols, fused masked max)
// All intermediates in LDS; weights from prepacked fragment buffers (L2).
// Per-wave tile loops use compile-time trip counts (8 waves exactly cover the
// tile grid) so no divergent exec-mask loops are generated.
// ---------------------------------------------------------------------------
__global__ __launch_bounds__(256)
void pointconv_kernel(const _Float16* __restrict__ xh,
                      const float* __restrict__ pos,
                      const float* __restrict__ new_pos,
                      const int* __restrict__ nidx,
                      const int* __restrict__ nvalid,
                      const _Float16* __restrict__ Wc1p, const float* __restrict__ bc1,
                      const _Float16* __restrict__ Wc2p, const float* __restrict__ bc2,
                      const _Float16* __restrict__ Wc3p, const float* __restrict__ bc3,
                      _Float16* __restrict__ encA) {
    __shared__ __align__(16) _Float16 feat[128 * 96];
    __shared__ __align__(16) _Float16 h1[128 * 64];
    __shared__ __align__(16) _Float16 h2[128 * 128];

    int bs = blockIdx.x;            // b*256 + s
    int b = bs >> 8;
    int tid = threadIdx.x, lane = tid & 31, wave = tid >> 5;
    int nv = nvalid[bs];
    float cx = new_pos[(size_t)bs * 3 + 0];
    float cy = new_pos[(size_t)bs * 3 + 1];
    float cz = new_pos[(size_t)bs * 3 + 2];

    // ---- build feat [k][0:64]=x_j, [64:67]=pos_j-pos_i, [67:96]=0
    {
        int k = tid >> 1;
        int half = tid & 1;
        int j = nidx[(size_t)bs * KNB + k];
        bool ok = k < nv;
        _Float16* dst = feat + k * 96 + half * 32;
        if (ok) {
            const _Float16* xr = xh + ((size_t)b * NN + j) * 64 + half * 32;
#pragma unroll
            for (int t = 0; t < 4; ++t) ((v8h*)dst)[t] = ((const v8h*)xr)[t];
        } else {
            for (int t = 0; t < 32; ++t) dst[t] = (_Float16)0.0f;
        }
        if (half == 1) {
            const float* pj = pos + ((size_t)b * NN + j) * 3;
            float dx = ok ? pj[0] - cx : 0.0f;
            float dy = ok ? pj[1] - cy : 0.0f;
            float dz = ok ? pj[2] - cz : 0.0f;
            _Float16* fr = feat + k * 96;
            fr[64] = (_Float16)dx; fr[65] = (_Float16)dy; fr[66] = (_Float16)dz;
            for (int t = 67; t < 96; ++t) fr[t] = (_Float16)0.0f;
        }
    }
    __syncthreads();

    int r16 = lane & 15;
    int kbo = (lane >> 4) << 3;
    int rowAdd = (lane >> 4) << 3;

    // ---- GEMM1: [128x96] x [96x64] (3 ktiles, 8x4 tile grid), lrelu -> h1
#pragma unroll
    for (int it = 0; it < 4; ++it) {
        int t = wave + it * 8;                 // 8 waves x 4 -> 32 tiles
        int rt = t >> 2, nt = t & 3;
        v8f acc = zero_v8f();
        const _Float16* Arow = feat + (rt * 16 + r16) * 96 + kbo;
#pragma unroll
        for (int kt = 0; kt < 3; ++kt) {
            v16h a = load_afrag(Arow + kt * 32);
            v16h bf = *(const v16h*)(Wc1p + (((size_t)kt * 4 + nt) * 32 + lane) * 16);
            acc = __builtin_amdgcn_wmma_f32_16x16x32_f16(false, a, false, bf,
                                                         (short)0, acc, false, false);
        }
        int col = nt * 16 + r16;
        float bv = bc1[col];
#pragma unroll
        for (int rr = 0; rr < 8; ++rr) {
            float v = acc[rr] + bv;
            v = v > 0.0f ? v : 0.01f * v;
            h1[(rt * 16 + rowAdd + rr) * 64 + col] = (_Float16)v;
        }
    }
    __syncthreads();

    // ---- GEMM2: [128x64] x [64x128] (2 ktiles, 8x8 tile grid), lrelu -> h2
#pragma unroll
    for (int it = 0; it < 8; ++it) {
        int t = wave + it * 8;                 // 8 waves x 8 -> 64 tiles
        int rt = t >> 3, nt = t & 7;
        v8f acc = zero_v8f();
        const _Float16* Arow = h1 + (rt * 16 + r16) * 64 + kbo;
#pragma unroll
        for (int kt = 0; kt < 2; ++kt) {
            v16h a = load_afrag(Arow + kt * 32);
            v16h bf = *(const v16h*)(Wc2p + (((size_t)kt * 8 + nt) * 32 + lane) * 16);
            acc = __builtin_amdgcn_wmma_f32_16x16x32_f16(false, a, false, bf,
                                                         (short)0, acc, false, false);
        }
        int col = nt * 16 + r16;
        float bv = bc2[col];
#pragma unroll
        for (int rr = 0; rr < 8; ++rr) {
            float v = acc[rr] + bv;
            v = v > 0.0f ? v : 0.01f * v;
            h2[(rt * 16 + rowAdd + rr) * 128 + col] = (_Float16)v;
        }
    }
    __syncthreads();

    // ---- GEMM3 + masked max over K: [128x128] x [128x512] (4 ktiles, 32 ctiles)
#pragma unroll
    for (int it = 0; it < 4; ++it) {
        int ct = wave + it * 8;                // 8 waves x 4 -> 32 col tiles
        float best = -3.402823466e38f;
        float bv = bc3[ct * 16 + r16];
        for (int rt = 0; rt < 8; ++rt) {
            v8f acc = zero_v8f();
            const _Float16* Arow = h2 + (rt * 16 + r16) * 128 + kbo;
#pragma unroll
            for (int kt = 0; kt < 4; ++kt) {
                v16h a = load_afrag(Arow + kt * 32);
                v16h bf = *(const v16h*)(Wc3p + (((size_t)kt * 32 + ct) * 32 + lane) * 16);
                acc = __builtin_amdgcn_wmma_f32_16x16x32_f16(false, a, false, bf,
                                                             (short)0, acc, false, false);
            }
#pragma unroll
            for (int rr = 0; rr < 8; ++rr) {
                int k = rt * 16 + rowAdd + rr;
                if (k < nv) {
                    float v = acc[rr] + bv;
                    v = v > 0.0f ? v : 0.01f * v;
                    best = fmaxf(best, v);
                }
            }
        }
        float other = __shfl_xor(best, 16);
        best = fmaxf(best, other);
        if (lane < 16) encA[(size_t)bs * 544 + ct * 16 + lane] = (_Float16)best;
    }
}

// ---------------------------------------------------------------------------
// Fill encoder A-matrix tail: cols 512..514 = new_pos, 515..543 = 0
// ---------------------------------------------------------------------------
__global__ void encfill_kernel(const float* __restrict__ new_pos,
                               _Float16* __restrict__ encA) {
    int row = blockIdx.x * blockDim.x + threadIdx.x;
    if (row >= BB * SS) return;
    _Float16* r = encA + (size_t)row * 544;
    r[512] = (_Float16)new_pos[(size_t)row * 3 + 0];
    r[513] = (_Float16)new_pos[(size_t)row * 3 + 1];
    r[514] = (_Float16)new_pos[(size_t)row * 3 + 2];
    for (int c = 515; c < 544; ++c) r[c] = (_Float16)0.0f;
}

// ---------------------------------------------------------------------------
// Generic WMMA GEMM: out = act(A[rows x Kp] * Wpacked + bias)
//   8 waves per WG; wave = one 16-row tile, up to NT(<=4) 16-col tiles.
//   Next A k-chunk is prefetched (global_prefetch) under the WMMAs.
// ---------------------------------------------------------------------------
__global__ __launch_bounds__(256)
void gemm_kernel(const _Float16* __restrict__ A, const _Float16* __restrict__ Wp,
                 const float* __restrict__ bias, int biasN,
                 _Float16* __restrict__ outH, float* __restrict__ outF,
                 int rows, int Kp, int Np, int act, int NT) {
    int lane = threadIdx.x & 31;
    int wave = threadIdx.x >> 5;
    int rowTile = blockIdx.x * 8 + wave;
    if (rowTile * 16 >= rows) return;
    int ntile0 = blockIdx.y * NT;
    int nTiles = Np >> 4;
    int kTiles = Kp >> 5;
    v8f acc[4];
#pragma unroll
    for (int i = 0; i < 4; ++i) acc[i] = zero_v8f();

    int r16 = lane & 15;
    int kbo = (lane >> 4) << 3;
    const _Float16* Arow = A + (size_t)(rowTile * 16 + r16) * Kp + kbo;

#pragma unroll 2
    for (int kt = 0; kt < kTiles; ++kt) {
        v16h a = load_afrag(Arow + kt * 32);
        if (kt + 1 < kTiles)
            __builtin_prefetch(Arow + (kt + 1) * 32, 0, 0);
#pragma unroll
        for (int i = 0; i < 4; ++i) {
            if (i < NT) {
                const _Float16* bp =
                    Wp + (((size_t)kt * nTiles + (ntile0 + i)) * 32 + lane) * 16;
                v16h bf = *(const v16h*)bp;
                acc[i] = __builtin_amdgcn_wmma_f32_16x16x32_f16(
                    false, a, false, bf, (short)0, acc[i], false, false);
            }
        }
    }
    int rowBase = rowTile * 16 + ((lane >> 4) << 3);
#pragma unroll
    for (int i = 0; i < 4; ++i) {
        if (i < NT) {
            int col = (ntile0 + i) * 16 + r16;
            float bv = (col < biasN) ? bias[col] : 0.0f;
#pragma unroll
            for (int rr = 0; rr < 8; ++rr) {
                int row = rowBase + rr;
                if (row < rows) {
                    float v = act_apply(acc[i][rr] + bv, act);
                    if (outH) outH[(size_t)row * Np + col] = (_Float16)v;
                    else      outF[(size_t)row * Np + col] = v;
                }
            }
        }
    }
}

// ---------------------------------------------------------------------------
// Vote aggregation: z = sum(mean/std)/sum(1/std), std = exp(0.5*logvar)
// e2 rows: [B*S, 2048]  (mean = [:,0:1024], logvar = [:,1024:2048])
// ---------------------------------------------------------------------------
__global__ void vote_kernel(const float* __restrict__ e2, float* __restrict__ z,
                            _Float16* __restrict__ zh) {
    int i = blockIdx.x * blockDim.x + threadIdx.x;
    if (i >= BB * BOT) return;
    int b = i >> 10;
    int c = i & 1023;
    const float* base = e2 + (size_t)b * SS * 2048;
    float num = 0.0f, den = 0.0f;
    for (int s = 0; s < SS; ++s) {
        float mean = base[(size_t)s * 2048 + c];
        float logv = base[(size_t)s * 2048 + 1024 + c];
        float inv = expf(-0.5f * logv);   // 1/std
        num += mean * inv;
        den += inv;
    }
    float zz = num / den;
    z[i] = zz;
    zh[i] = (_Float16)zz;
}

// ---------------------------------------------------------------------------
// Build fold1 A-matrix: [code(1024) | grid(2) | 0...] per (b,g) row (f16)
// ---------------------------------------------------------------------------
__global__ void fold1_build(const _Float16* __restrict__ zh,
                            _Float16* __restrict__ foldA) {
    int idx = blockIdx.x * blockDim.x + threadIdx.x;   // row*33 + chunk
    int row = idx / 33;
    int ch = idx % 33;
    if (row >= ROWS_DEC) return;
    _Float16* dst = foldA + (size_t)row * 1056 + ch * 32;
    if (row >= BB * GG) {
        for (int t = 0; t < 32; ++t) dst[t] = (_Float16)0.0f;
        return;
    }
    int b = row / GG;
    int g = row % GG;
    if (ch < 32) {
        const _Float16* src = zh + (size_t)b * BOT + ch * 32;
#pragma unroll
        for (int t = 0; t < 4; ++t) ((v8h*)dst)[t] = ((const v8h*)src)[t];
    } else {
        float gx = -0.3f + 0.6f * (float)(g % MG) / (float)(MG - 1);
        float gy = -0.3f + 0.6f * (float)(g / MG) / (float)(MG - 1);
        dst[0] = (_Float16)gx;
        dst[1] = (_Float16)gy;
        for (int t = 2; t < 32; ++t) dst[t] = (_Float16)0.0f;
    }
}

// ---------------------------------------------------------------------------
// Patch fold2 A-matrix: cols 1024..1026 = f (first fold output)
// ---------------------------------------------------------------------------
__global__ void fold2_update(const float* __restrict__ f3,
                             _Float16* __restrict__ foldA) {
    int row = blockIdx.x * blockDim.x + threadIdx.x;
    if (row >= BB * GG) return;
    _Float16* dst = foldA + (size_t)row * 1056 + 1024;
    dst[0] = (_Float16)f3[(size_t)row * 16 + 0];
    dst[1] = (_Float16)f3[(size_t)row * 16 + 1];
    dst[2] = (_Float16)f3[(size_t)row * 16 + 2];
}

__global__ void outcopy_kernel(const float* __restrict__ f3, float* __restrict__ out) {
    int i = blockIdx.x * blockDim.x + threadIdx.x;
    if (i >= BB * GG * 3) return;
    int row = i / 3;
    int c = i % 3;
    out[i] = f3[(size_t)row * 16 + c];
}

// ---------------------------------------------------------------------------
// Host launcher
// ---------------------------------------------------------------------------
extern "C" void kernel_launch(void* const* d_in, const int* in_sizes, int n_in,
                              void* d_out, int out_size, void* d_ws, size_t ws_size,
                              hipStream_t stream) {
    (void)in_sizes; (void)n_in; (void)out_size; (void)ws_size;

    const float* pos  = (const float*)d_in[0];
    const float* Wt   = (const float*)d_in[1];
    const float* bt   = (const float*)d_in[2];
    const float* Wc1  = (const float*)d_in[3];
    const float* bc1  = (const float*)d_in[4];
    const float* Wc2  = (const float*)d_in[5];
    const float* bc2  = (const float*)d_in[6];
    const float* Wc3  = (const float*)d_in[7];
    const float* bc3  = (const float*)d_in[8];
    const float* We1  = (const float*)d_in[9];
    const float* be1  = (const float*)d_in[10];
    const float* We2  = (const float*)d_in[11];
    const float* be2  = (const float*)d_in[12];
    const float* Wf11 = (const float*)d_in[13];
    const float* bf11 = (const float*)d_in[14];
    const float* Wf12 = (const float*)d_in[15];
    const float* bf12 = (const float*)d_in[16];
    const float* Wf13 = (const float*)d_in[17];
    const float* bf13 = (const float*)d_in[18];
    const float* Wf21 = (const float*)d_in[19];
    const float* bf21 = (const float*)d_in[20];
    const float* Wf22 = (const float*)d_in[21];
    const float* bf22 = (const float*)d_in[22];
    const float* Wf23 = (const float*)d_in[23];
    const float* bf23 = (const float*)d_in[24];
    float* out = (float*)d_out;

    char* ws = (char*)d_ws;
    size_t off = 0;
    auto carve = [&](size_t bytes) -> char* {
        char* p = ws + off;
        off = (off + bytes + 255) & ~(size_t)255;
        return p;
    };

    _Float16* xh    = (_Float16*)carve((size_t)BB * NN * 64 * 2);
    int*      fpsI  = (int*)carve((size_t)BB * SS * 4);
    float*    npos  = (float*)carve((size_t)BB * SS * 3 * 4);
    int*      nidx  = (int*)carve((size_t)BB * SS * KNB * 4);
    int*      nval  = (int*)carve((size_t)BB * SS * 4);
    _Float16* encA  = (_Float16*)carve((size_t)BB * SS * 544 * 2);
    _Float16* e1    = (_Float16*)carve((size_t)BB * SS * 512 * 2);
    float*    e2    = (float*)carve((size_t)BB * SS * 2048 * 4);
    float*    z     = (float*)carve((size_t)BB * BOT * 4);
    _Float16* zh    = (_Float16*)carve((size_t)BB * BOT * 2);
    _Float16* foldA = (_Float16*)carve((size_t)ROWS_DEC * 1056 * 2);
    _Float16* dd1   = (_Float16*)carve((size_t)ROWS_DEC * 512 * 2);
    _Float16* dd2   = (_Float16*)carve((size_t)ROWS_DEC * 512 * 2);
    float*    f3    = (float*)carve((size_t)ROWS_DEC * 16 * 4);
    _Float16* Wc1p  = (_Float16*)carve((size_t)96 * 64 * 2);
    _Float16* Wc2p  = (_Float16*)carve((size_t)64 * 128 * 2);
    _Float16* Wc3p  = (_Float16*)carve((size_t)128 * 512 * 2);
    _Float16* We1p  = (_Float16*)carve((size_t)544 * 512 * 2);
    _Float16* We2p  = (_Float16*)carve((size_t)512 * 2048 * 2);
    _Float16* Wf11p = (_Float16*)carve((size_t)1056 * 512 * 2);
    _Float16* Wf12p = (_Float16*)carve((size_t)512 * 512 * 2);
    _Float16* Wf13p = (_Float16*)carve((size_t)512 * 16 * 2);
    _Float16* Wf21p = (_Float16*)carve((size_t)1056 * 512 * 2);
    _Float16* Wf22p = (_Float16*)carve((size_t)512 * 512 * 2);
    _Float16* Wf23p = (_Float16*)carve((size_t)512 * 16 * 2);

    auto pack = [&](const float* W, int K, int Nn, int Kp, int Np, _Float16* dst) {
        int total = Kp * Np;
        packw_kernel<<<(total + 255) / 256, 256, 0, stream>>>(W, K, Nn, Kp, Np, dst);
    };
    pack(Wc1,  67,   64,   96,   64,   Wc1p);
    pack(Wc2,  64,   128,  64,   128,  Wc2p);
    pack(Wc3,  128,  512,  128,  512,  Wc3p);
    pack(We1,  515,  512,  544,  512,  We1p);
    pack(We2,  512,  2048, 512,  2048, We2p);
    pack(Wf11, 1026, 512,  1056, 512,  Wf11p);
    pack(Wf12, 512,  512,  512,  512,  Wf12p);
    pack(Wf13, 512,  3,    512,  16,   Wf13p);
    pack(Wf21, 1027, 512,  1056, 512,  Wf21p);
    pack(Wf22, 512,  512,  512,  512,  Wf22p);
    pack(Wf23, 512,  3,    512,  16,   Wf23p);

    auto gemm = [&](const _Float16* A, const _Float16* Wp, const float* bias,
                    int biasN, _Float16* oH, float* oF, int rows, int Kp, int Np,
                    int act) {
        int nTiles = Np / 16;
        int NT = (nTiles % 4 == 0) ? 4 : 1;
        dim3 grid((rows / 16 + 7) / 8, (nTiles + NT - 1) / NT);
        gemm_kernel<<<grid, 256, 0, stream>>>(A, Wp, bias, biasN, oH, oF, rows, Kp,
                                              Np, act, NT);
    };

    // Stage 1: transformer MLP
    transformer_kernel<<<(BB * NN * 64 + 255) / 256, 256, 0, stream>>>(pos, Wt, bt, xh);

    // Stage 2: FPS (one block per cloud)
    fps_kernel<<<BB, 1024, 0, stream>>>(pos, fpsI, npos);

    // Stage 3: radius ball query (one wave per sample)
    ballq_kernel<<<(BB * SS * 32 + 255) / 256, 256, 0, stream>>>(pos, npos, nidx, nval);

    // Stage 4: fused PointConv MLP + masked max aggregation -> encA[:, 0:512]
    pointconv_kernel<<<BB * SS, 256, 0, stream>>>(xh, pos, npos, nidx, nval, Wc1p,
                                                  bc1, Wc2p, bc2, Wc3p, bc3, encA);
    encfill_kernel<<<(BB * SS + 255) / 256, 256, 0, stream>>>(npos, encA);

    // Stage 5: encoder
    gemm(encA, We1p, be1, 512, e1, nullptr, BB * SS, 544, 512, /*lrelu*/ 1);
    gemm(e1, We2p, be2, 2048, nullptr, e2, BB * SS, 512, 2048, /*none*/ 0);

    // Stage 6: vote aggregation
    vote_kernel<<<(BB * BOT + 255) / 256, 256, 0, stream>>>(e2, z, zh);

    // Stage 7: FoldingNet decoder, fold 1
    {
        int total = ROWS_DEC * 33;
        fold1_build<<<(total + 255) / 256, 256, 0, stream>>>(zh, foldA);
    }
    gemm(foldA, Wf11p, bf11, 512, dd1, nullptr, ROWS_DEC, 1056, 512, /*relu*/ 2);
    gemm(dd1, Wf12p, bf12, 512, dd2, nullptr, ROWS_DEC, 512, 512, /*relu*/ 2);
    gemm(dd2, Wf13p, bf13, 3, nullptr, f3, ROWS_DEC, 512, 16, /*none*/ 0);

    // Stage 8: fold 2
    fold2_update<<<(BB * GG + 255) / 256, 256, 0, stream>>>(f3, foldA);
    gemm(foldA, Wf21p, bf21, 512, dd1, nullptr, ROWS_DEC, 1056, 512, /*relu*/ 2);
    gemm(dd1, Wf22p, bf22, 512, dd2, nullptr, ROWS_DEC, 512, 512, /*relu*/ 2);
    gemm(dd2, Wf23p, bf23, 3, nullptr, f3, ROWS_DEC, 512, 16, /*none*/ 0);

    // Final: gather [B,G,3] output
    outcopy_kernel<<<(BB * GG * 3 + 255) / 256, 256, 0, stream>>>(f3, out);
}
